// WindowCrossAttention_33397665693862
// MI455X (gfx1250) — matrix operand order repaired and checked by reading, not compiled, not run on hardware
//
#include <hip/hip_runtime.h>
#include <hip/hip_bf16.h>

#define DIM     1024
#define HEADS   16
#define HD      64
#define NTOK    49
#define NPAD    64
#define NWIN    1024
#define QK_SCALE 0.125f   // 64^-0.5

typedef __attribute__((ext_vector_type(16))) __bf16        v16bf;
typedef __attribute__((ext_vector_type(8)))  float          v8f;
typedef __attribute__((ext_vector_type(8)))  unsigned short v8us;

union Frag {
    v16bf v;
    v8us  h[2];
    unsigned short s[16];
};

__device__ __forceinline__ unsigned short f2bf(float f) {
    union { float f; unsigned int u; } x;
    x.f = f;
    unsigned int r = x.u + 0x7FFFu + ((x.u >> 16) & 1u);  // round-to-nearest-even
    return (unsigned short)(r >> 16);
}

__device__ __forceinline__ v8f wmma_bf16(v16bf a, v16bf b, v8f c) {
    // (neg_a, A, neg_b, B, c_mod, C, reuse_a, reuse_b)
    return __builtin_amdgcn_wmma_f32_16x16x32_bf16(false, a, false, b, (short)0, c,
                                                   false, false);
}

// A fragment (16x32 bf16): lane holds row (lane&15); 16-wide contiguous K run c0..c0+15,
// c0 = (lane>>4)*16.  rowptr -> start of this lane's row at the current 32-wide K chunk.
// Unconditional: all source buffers are zero-padded, so every row is loadable.
__device__ __forceinline__ v16bf ldA_bf16(const unsigned short* rowptr, int lane) {
    Frag u;
    const int c0 = (lane >> 4) * 16;
    u.h[0] = *(const v8us*)(rowptr + c0);
    u.h[1] = *(const v8us*)(rowptr + c0 + 8);
    return u.v;
}

// B fragment (32x16 bf16): lane holds column (lane&15); 16-wide contiguous K run at
// (lane>>4)*16.  colptr must already point at &B[col][kchunk + (lane>>4)*16].
__device__ __forceinline__ v16bf ldB_bf16(const unsigned short* colptr) {
    Frag u;
    u.h[0] = *(const v8us*)(colptr);
    u.h[1] = *(const v8us*)(colptr + 8);
    return u.v;
}

// ---------------------------------------------------------------------------
// Kernel 1a: convert the four 1024x1024 fp32 weight matrices to bf16.
// ---------------------------------------------------------------------------
__global__ __launch_bounds__(256) void cvt_w_kernel(
    const float* __restrict__ Wq, const float* __restrict__ Wk,
    const float* __restrict__ Wv, const float* __restrict__ Wo,
    unsigned short* __restrict__ Wb)
{
    const int t = blockIdx.y;  // 0..3
    const int i = blockIdx.x * 256 + threadIdx.x;
    const float* src = (t == 0) ? Wq : (t == 1) ? Wk : (t == 2) ? Wv : Wo;
    Wb[(size_t)t * DIM * DIM + i] = f2bf(src[i]);
}

// ---------------------------------------------------------------------------
// Kernel 1b: convert activations fp32 [b,49,1024] -> bf16 padded [b,64,1024],
// pad rows zero-filled.  8 elements per thread, vectorized 16B stores.
// ---------------------------------------------------------------------------
__global__ __launch_bounds__(256) void cvt_x_kernel(
    const float* __restrict__ lm, const float* __restrict__ img,
    unsigned short* __restrict__ Xlm, unsigned short* __restrict__ Ximg)
{
    const int which = blockIdx.y;  // 0 = landmark, 1 = image
    const float* src = which ? img : lm;
    unsigned short* dst = which ? Ximg : Xlm;

    const size_t e = ((size_t)blockIdx.x * 256 + threadIdx.x) * 8;  // element index
    const int b   = (int)(e >> 16);         // / (64*1024)
    const int row = (int)((e >> 10) & 63);  // padded row
    const int col = (int)(e & 1023);

    v8us v;
    if (row < NTOK) {
        const float* s = src + ((size_t)b * NTOK + row) * DIM + col;
#pragma unroll
        for (int i = 0; i < 8; ++i) v[i] = f2bf(s[i]);
    } else {
#pragma unroll
        for (int i = 0; i < 8; ++i) v[i] = 0;
    }
    *(v8us*)(dst + e) = v;
}

// ---------------------------------------------------------------------------
// Kernel 2: fused Q/K/V projection.  out = x @ W^T + b, pure bf16 data path.
// grid = (NWIN, 8 n-chunks of 128, 3 which); block = 256 (8 wave32).
// Q,K stored [b,h,tok,d]; V stored transposed [b,h,d,tok] for the PV GEMM.
// ---------------------------------------------------------------------------
__global__ __launch_bounds__(256) void qkv_proj_kernel(
    const unsigned short* __restrict__ Xlm, const unsigned short* __restrict__ Ximg,
    const unsigned short* __restrict__ Wb,
    const float* __restrict__ bq, const float* __restrict__ bk, const float* __restrict__ bv,
    unsigned short* __restrict__ Qb, unsigned short* __restrict__ Kb,
    unsigned short* __restrict__ Vtb)
{
    const int lane  = threadIdx.x & 31;
    const int wave  = threadIdx.x >> 5;
    const int b     = blockIdx.x;
    const int nblk  = blockIdx.y;      // 128 output features per block
    const int which = blockIdx.z;      // 0=Q 1=K 2=V

    const int mt    = wave & 3;                  // m-tile (rows mt*16 .. +15 of padded 64)
    const int ng    = wave >> 2;                 // n-group 0..1
    const int nbase = nblk * 128 + ng * 64;      // wave covers 4 n-tiles of 16

    const unsigned short* X = (which == 0) ? Xlm : Ximg;
    const unsigned short* W = Wb + (size_t)which * DIM * DIM;

    const int arow = mt * 16 + (lane & 15);
    const unsigned short* arowp = X + ((size_t)b * NPAD + arow) * DIM;

    v8f acc[4];
#pragma unroll
    for (int j = 0; j < 4; ++j)
#pragma unroll
        for (int r = 0; r < 8; ++r) acc[j][r] = 0.0f;

    for (int kc = 0; kc < DIM / 32; ++kc) {
        const int kbase = kc * 32;
        v16bf a = ldA_bf16(arowp + kbase, lane);
#pragma unroll
        for (int j = 0; j < 4; ++j) {
            const unsigned short* bp =
                W + (size_t)(nbase + j * 16 + (lane & 15)) * DIM + kbase + ((lane >> 4) * 16);
            acc[j] = wmma_bf16(a, ldB_bf16(bp), acc[j]);
        }
    }

    const float* bias = (which == 0) ? bq : (which == 1) ? bk : bv;
    unsigned short* dst = (which == 0) ? Qb : (which == 1) ? Kb : Vtb;

    const int m0 = mt * 16 + (lane >> 4) * 8;
#pragma unroll
    for (int j = 0; j < 4; ++j) {
        const int nf = nbase + j * 16 + (lane & 15);
        const int h  = nf >> 6;
        const int d  = nf & 63;
        const float bias_v = bias[nf];
        const size_t base = ((size_t)b * HEADS + h) * (NPAD * HD);
#pragma unroll
        for (int r = 0; r < 8; ++r) {
            const int m = m0 + r;
            const unsigned short v = f2bf(acc[j][r] + bias_v);
            if (which <= 1) dst[base + (size_t)m * HD + d] = v;
            else            dst[base + (size_t)d * NPAD + m] = v;   // V transposed
        }
    }
}

// ---------------------------------------------------------------------------
// Kernel 3: attention per (b,h).  One wave32 per block; 24 KB LDS.
// S = Q K^T (32 WMMA) -> LDS; softmax + rel-pos bias + pad mask -> bf16 P in LDS;
// O = P V (32 WMMA) -> ctx[b, tok, h*64+d] bf16 (all 64 rows stored).
// ---------------------------------------------------------------------------
__global__ __launch_bounds__(32) void attn_kernel(
    const unsigned short* __restrict__ Qb, const unsigned short* __restrict__ Kb,
    const unsigned short* __restrict__ Vtb,
    const float* __restrict__ bias_table, const int* __restrict__ rel_idx,
    unsigned short* __restrict__ ctx)
{
    const int lane = threadIdx.x;
    const int bh   = blockIdx.x;
    const int b    = bh >> 4;
    const int h    = bh & 15;

    const unsigned short* Q  = Qb  + (size_t)bh * (NPAD * HD);
    const unsigned short* K  = Kb  + (size_t)bh * (NPAD * HD);
    const unsigned short* Vt = Vtb + (size_t)bh * (NPAD * HD);

    __shared__ __align__(16) float          S[NPAD * NPAD];   // 16 KB
    __shared__ __align__(16) unsigned short P[NPAD * NPAD];   //  8 KB

    // ---- S = Q K^T ----
    v16bf kb[2][4];
#pragma unroll
    for (int kk = 0; kk < 2; ++kk)
#pragma unroll
        for (int nt = 0; nt < 4; ++nt) {
            const int keytok = nt * 16 + (lane & 15);
            kb[kk][nt] = ldB_bf16(K + keytok * HD + kk * 32 + ((lane >> 4) * 16));
        }

#pragma unroll
    for (int mt = 0; mt < 4; ++mt) {
        v16bf qa[2];
        const int qrow = mt * 16 + (lane & 15);
#pragma unroll
        for (int kk = 0; kk < 2; ++kk)
            qa[kk] = ldA_bf16(Q + qrow * HD + kk * 32, lane);
#pragma unroll
        for (int nt = 0; nt < 4; ++nt) {
            v8f acc;
#pragma unroll
            for (int r = 0; r < 8; ++r) acc[r] = 0.0f;
#pragma unroll
            for (int kk = 0; kk < 2; ++kk) acc = wmma_bf16(qa[kk], kb[kk][nt], acc);
            const int m0 = mt * 16 + (lane >> 4) * 8;
            const int n  = nt * 16 + (lane & 15);
#pragma unroll
            for (int r = 0; r < 8; ++r) S[(m0 + r) * NPAD + n] = acc[r];
        }
    }
    __syncthreads();

    // ---- softmax rows (2 rows per lane), with bias gather + pad masking ----
    for (int q = lane; q < NPAD; q += 32) {
        const int qc = (q < NTOK) ? q : 0;
        float mx = -1e30f;
        for (int k = 0; k < NTOK; ++k) {
            float s = S[q * NPAD + k] * QK_SCALE
                    + bias_table[rel_idx[qc * NTOK + k] * HEADS + h];
            S[q * NPAD + k] = s;
            mx = fmaxf(mx, s);
        }
        float sum = 0.0f;
        for (int k = 0; k < NTOK; ++k) {
            float e = __expf(S[q * NPAD + k] - mx);
            S[q * NPAD + k] = e;
            sum += e;
        }
        const float inv = 1.0f / sum;
        for (int k = 0; k < NTOK; ++k) P[q * NPAD + k] = f2bf(S[q * NPAD + k] * inv);
        for (int k = NTOK; k < NPAD; ++k) P[q * NPAD + k] = 0;  // mask padded keys
    }
    __syncthreads();

    // ---- O = P V  (V stored transposed: column d is contiguous over tokens) ----
    v16bf vb[2][4];
#pragma unroll
    for (int kk = 0; kk < 2; ++kk)
#pragma unroll
        for (int nt = 0; nt < 4; ++nt) {
            const int d = nt * 16 + (lane & 15);
            vb[kk][nt] = ldB_bf16(Vt + d * NPAD + kk * 32 + ((lane >> 4) * 16));
        }

#pragma unroll
    for (int mt = 0; mt < 4; ++mt) {
        v16bf pa[2];
        const int qrow = mt * 16 + (lane & 15);
#pragma unroll
        for (int kk = 0; kk < 2; ++kk)
            pa[kk] = ldA_bf16(P + qrow * NPAD + kk * 32, lane);  // ds_load_b128
#pragma unroll
        for (int nt = 0; nt < 4; ++nt) {
            v8f acc;
#pragma unroll
            for (int r = 0; r < 8; ++r) acc[r] = 0.0f;
#pragma unroll
            for (int kk = 0; kk < 2; ++kk) acc = wmma_bf16(pa[kk], vb[kk][nt], acc);
            const int m0 = mt * 16 + (lane >> 4) * 8;
            const int n  = h * HD + nt * 16 + (lane & 15);
#pragma unroll
            for (int r = 0; r < 8; ++r) {
                const int m = m0 + r;  // store all rows; pad rows are deterministic
                ctx[((size_t)b * NPAD + m) * DIM + n] = f2bf(acc[r]);
            }
        }
    }
}

// ---------------------------------------------------------------------------
// Kernel 4: output projection.  out = ctx @ Wo^T + bo, fp32 result, valid rows only.
// ---------------------------------------------------------------------------
__global__ __launch_bounds__(256) void out_proj_kernel(
    const unsigned short* __restrict__ ctx, const unsigned short* __restrict__ Wo,
    const float* __restrict__ bo, float* __restrict__ out)
{
    const int lane  = threadIdx.x & 31;
    const int wave  = threadIdx.x >> 5;
    const int b     = blockIdx.x;
    const int nblk  = blockIdx.y;
    const int mt    = wave & 3;
    const int ng    = wave >> 2;
    const int nbase = nblk * 128 + ng * 64;

    const int arow = mt * 16 + (lane & 15);
    const unsigned short* arowp = ctx + ((size_t)b * NPAD + arow) * DIM;

    v8f acc[4];
#pragma unroll
    for (int j = 0; j < 4; ++j)
#pragma unroll
        for (int r = 0; r < 8; ++r) acc[j][r] = 0.0f;

    for (int kc = 0; kc < DIM / 32; ++kc) {
        const int kbase = kc * 32;
        v16bf a = ldA_bf16(arowp + kbase, lane);
#pragma unroll
        for (int j = 0; j < 4; ++j) {
            const unsigned short* bp =
                Wo + (size_t)(nbase + j * 16 + (lane & 15)) * DIM + kbase + ((lane >> 4) * 16);
            acc[j] = wmma_bf16(a, ldB_bf16(bp), acc[j]);
        }
    }

    const int m0 = mt * 16 + (lane >> 4) * 8;
#pragma unroll
    for (int j = 0; j < 4; ++j) {
        const int nf = nbase + j * 16 + (lane & 15);
        const float bias_v = bo[nf];
#pragma unroll
        for (int r = 0; r < 8; ++r) {
            const int m = m0 + r;
            if (m < NTOK)
                out[((size_t)b * NTOK + m) * DIM + nf] = acc[j][r] + bias_v;
        }
    }
}

// ---------------------------------------------------------------------------
extern "C" void kernel_launch(void* const* d_in, const int* in_sizes, int n_in,
                              void* d_out, int out_size, void* d_ws, size_t ws_size,
                              hipStream_t stream) {
    const float* lm         = (const float*)d_in[0];
    const float* img        = (const float*)d_in[1];
    const float* Wq         = (const float*)d_in[2];
    const float* bq         = (const float*)d_in[3];
    const float* Wk         = (const float*)d_in[4];
    const float* bk         = (const float*)d_in[5];
    const float* Wv         = (const float*)d_in[6];
    const float* bv         = (const float*)d_in[7];
    const float* Wo         = (const float*)d_in[8];
    const float* bo         = (const float*)d_in[9];
    const float* bias_table = (const float*)d_in[10];
    const int*   rel_idx    = (const int*)d_in[11];
    float*       out        = (float*)d_out;

    // Workspace layout (ushort elements):
    //   Wb   : 4 * 1024*1024        bf16 weights (8 MB, L2-resident)
    //   Xlm  : 1024*64*1024         bf16 landmark, padded  (134 MB) -- reused as ctx
    //   Ximg : 1024*64*1024         bf16 image, padded     (134 MB)
    //   Qb   : 1024*16*64*64        bf16 Q   [b,h,tok,d]   (134 MB)
    //   Kb   : 1024*16*64*64        bf16 K   [b,h,tok,d]   (134 MB)
    //   Vtb  : 1024*16*64*64        bf16 V^T [b,h,d,tok]   (134 MB)
    unsigned short* Wb   = (unsigned short*)d_ws;
    const size_t wsz     = (size_t)4 * DIM * DIM;
    const size_t xsz     = (size_t)NWIN * NPAD * DIM;          // == NWIN*HEADS*NPAD*HD
    unsigned short* Xlm  = Wb   + wsz;
    unsigned short* Ximg = Xlm  + xsz;
    unsigned short* Qb   = Ximg + xsz;
    unsigned short* Kb   = Qb   + xsz;
    unsigned short* Vtb  = Kb   + xsz;
    unsigned short* ctx  = Xlm;  // Xlm is dead after qkv_proj; alias saves 134 MB

    {   // weights -> bf16
        dim3 grid(DIM * DIM / 256, 4);
        cvt_w_kernel<<<grid, 256, 0, stream>>>(Wq, Wk, Wv, Wo, Wb);
    }
    {   // activations -> bf16, zero-padded to 64 rows/window
        dim3 grid((unsigned)(xsz / 8 / 256), 2);
        cvt_x_kernel<<<grid, 256, 0, stream>>>(lm, img, Xlm, Ximg);
    }
    {   // fused QKV projection
        dim3 grid(NWIN, DIM / 128, 3);
        qkv_proj_kernel<<<grid, 256, 0, stream>>>(Xlm, Ximg, Wb, bq, bk, bv, Qb, Kb, Vtb);
    }
    {   // windowed attention, one wave per (b,h)
        attn_kernel<<<NWIN * HEADS, 32, 0, stream>>>(Qb, Kb, Vtb, bias_table, rel_idx, ctx);
    }
    {   // output projection
        dim3 grid(NWIN, DIM / 128);
        out_proj_kernel<<<grid, 256, 0, stream>>>(ctx, Wb + 3 * (size_t)DIM * DIM, bo, out);
    }
}
